// TriangleAttention_26637387170081
// MI455X (gfx1250) — matrix operand order repaired and checked by reading, not compiled, not run on hardware
//
#include <hip/hip_runtime.h>
#include <math.h>

typedef unsigned short ushort_t;
typedef unsigned int   uint_t;
typedef __attribute__((ext_vector_type(16))) __bf16        v16bf;
typedef __attribute__((ext_vector_type(8)))  __bf16        bf16x8;
typedef __attribute__((ext_vector_type(8)))  float         v8f;
typedef __attribute__((ext_vector_type(8)))  unsigned short u16x8;

#define S_DIM 512
#define D_DIM 128
#define H_DIM 4
#define HD_DIM 32
#define NB 64            // S / 8 blocks per dimension
#define NPAIR 2080       // NB*(NB+1)/2
#define XSTR 136         // pair / attn-out tile stride (bf16 elems), 272B rows (16B aligned)
#define QSTR 392         // qkv tile stride (bf16 elems), 784B rows (16B aligned)
#define X_ELEMS (128 * XSTR)
#define LDS_USHORTS (X_ELEMS + 128 * QSTR)   // 67584 ushorts = 135168 bytes
#define ATT_SCALE 0.17677669529663687f       // 32^-0.5

// Native conversions: let the compiler use hardware bf16 converts (packed where possible).
__device__ __forceinline__ ushort_t f2bf(float f) {
    __bf16 h = (__bf16)f;
    return __builtin_bit_cast(ushort_t, h);
}
__device__ __forceinline__ float bf2f(ushort_t u) {
    return (float)__builtin_bit_cast(__bf16, u);
}

// A operand 16x32 bf16 from LDS.  ISA layout: lanes 0-15: M=lane, K = k0+0..7 (v0..3)
// and k0+16..23 (v4..7); lanes 16-31: M=lane-16, K = k0+8..15 and k0+24..31.
__device__ __forceinline__ v16bf load_a_lds(const ushort_t* lds, int m0, int k0, int stride) {
    int lane = threadIdx.x & 31;
    int half = lane >> 4;
    int m = m0 + (lane & 15);
    const ushort_t* p = lds + m * stride + k0 + half * 8;
    bf16x8 lo = *(const bf16x8*)p;          // K = k0 + half*8 + 0..7
    bf16x8 hi = *(const bf16x8*)(p + 16);   // K = k0 + 16 + half*8 + 0..7
    return __builtin_shufflevector(lo, hi, 0,1,2,3,4,5,6,7,8,9,10,11,12,13,14,15);
}

// B operand 32x16 bf16 straight from global (weights stored row-major [N][K], K-stride 128).
// ISA layout: lanes 0-15: N=lane, K=k0+0..7 and k0+16..23; lanes 16-31: N=lane-16, K=k0+8..15 and k0+24..31.
__device__ __forceinline__ v16bf load_b_glb(const ushort_t* __restrict__ w, int n0, int k0) {
    int lane = threadIdx.x & 31;
    int half = lane >> 4;
    int n = n0 + (lane & 15);
    const ushort_t* p = w + n * D_DIM + k0 + half * 8;
    bf16x8 lo = *(const bf16x8*)p;
    bf16x8 hi = *(const bf16x8*)(p + 16);
    return __builtin_shufflevector(lo, hi, 0,1,2,3,4,5,6,7,8,9,10,11,12,13,14,15);
}

__device__ __forceinline__ v8f wmma_bf16(v16bf a, v16bf b, v8f c) {
    return __builtin_amdgcn_wmma_f32_16x16x32_bf16(
        /*neg_a=*/false, a, /*neg_b=*/false, b,
        /*c_mod=*/(short)0, c, /*reuse_a=*/false, /*reuse_b=*/false);
}

// Convert W_qkv (384x128) and W_out (128x128) f32 -> bf16 into workspace.
__global__ void __launch_bounds__(256) tri_convert_weights(
    const float* __restrict__ wq, const float* __restrict__ wo, ushort_t* __restrict__ dst) {
    int i = blockIdx.x * 256 + threadIdx.x;
    if (i < 3 * D_DIM * D_DIM) dst[i] = f2bf(wq[i]);
    if (i < D_DIM * D_DIM)     dst[3 * D_DIM * D_DIM + i] = f2bf(wo[i]);
}

__global__ void __launch_bounds__(256) tri_attn_kernel(
    const float* __restrict__ pair,
    const ushort_t* __restrict__ wq_bf,
    const ushort_t* __restrict__ wo_bf,
    const float* __restrict__ b_out,
    float* __restrict__ out) {
    extern __shared__ ushort_t smem[];
    ushort_t* lds_x   = smem;             // 128 x XSTR : pair tile, later attn-out tile
    ushort_t* lds_qkv = smem + X_ELEMS;   // 128 x QSTR : qkv (bf16)

    // ---- decode (b, bi, bj) from triangular pair index ----
    int wg = blockIdx.x;
    int b  = wg / NPAIR;
    int pr = wg - b * NPAIR;
    int bi = (int)((sqrtf((float)(8 * pr + 1)) - 1.0f) * 0.5f);
    while ((bi + 1) * (bi + 2) / 2 <= pr) bi++;
    while (bi * (bi + 1) / 2 > pr) bi--;
    int bj = pr - bi * (bi + 1) / 2;
    int i0 = bi * 8, j0 = bj * 8;
    bool diag = (bi == bj);
    int tid = threadIdx.x;

    // ---- phase 0: pair tile -> LDS (bf16). rows 0-63 = block (bi,bj), 64-127 = (bj,bi) ----
    {
        int row   = tid >> 1;           // 0..127 (two threads per row)
        int halfc = (tid & 1) * 64;     // column offset within D
        int r = (row >> 3) & 7, c = row & 7;
        int pi = (row < 64) ? (i0 + r) : (j0 + r);
        int pj = (row < 64) ? (j0 + c) : (i0 + c);
        const float4* src = (const float4*)(pair + (((size_t)(b * S_DIM + pi)) * S_DIM + pj) * D_DIM + halfc);
        ushort_t* dst = lds_x + row * XSTR + halfc;
#pragma unroll
        for (int i = 0; i < 16; i++) {
            float4 v = src[i];
            uint_t w0 = (uint_t)f2bf(v.x) | ((uint_t)f2bf(v.y) << 16);
            uint_t w1 = (uint_t)f2bf(v.z) | ((uint_t)f2bf(v.w) << 16);
            *(uint2*)(dst + i * 4) = make_uint2(w0, w1);
        }
    }
    __syncthreads();

    int wave  = tid >> 5;
    int lane  = tid & 31;
    int halfw = lane >> 4;
    int laneN = lane & 15;

    // ---- phase 1: QKV GEMM  (M=128, N=384, K=128), wave w owns M-tile w ----
    {
        v16bf a0 = load_a_lds(lds_x, wave * 16, 0,  XSTR);
        v16bf a1 = load_a_lds(lds_x, wave * 16, 32, XSTR);
        v16bf a2 = load_a_lds(lds_x, wave * 16, 64, XSTR);
        v16bf a3 = load_a_lds(lds_x, wave * 16, 96, XSTR);
#pragma unroll 2
        for (int nt = 0; nt < 24; nt++) {
            v8f c = {};
            c = wmma_bf16(a0, load_b_glb(wq_bf, nt * 16, 0),  c);
            c = wmma_bf16(a1, load_b_glb(wq_bf, nt * 16, 32), c);
            c = wmma_bf16(a2, load_b_glb(wq_bf, nt * 16, 64), c);
            c = wmma_bf16(a3, load_b_glb(wq_bf, nt * 16, 96), c);
            int n = nt * 16 + laneN;
#pragma unroll
            for (int r = 0; r < 8; r++)   // C layout: M = r + 8*halfw, N = laneN
                lds_qkv[(wave * 16 + halfw * 8 + r) * QSTR + n] = f2bf(c[r]);
        }
    }
    __syncthreads();

    // ---- phase 2: per-(pixel, head) attention; q from self, k/v from transposed partner ----
    for (int task = tid; task < 128 * H_DIM; task += 256) {
        int p = task >> 2, h = task & 3;
        int r = (p >> 3) & 7, c = p & 7;
        int partner = (p < 64) ? (64 + c * 8 + r) : (c * 8 + r);
        const u16x8* q8 = (const u16x8*)(lds_qkv + p * QSTR + h * HD_DIM);
        const u16x8* k8 = (const u16x8*)(lds_qkv + partner * QSTR + D_DIM);
        const u16x8* v8 = (const u16x8*)(lds_qkv + partner * QSTR + 2 * D_DIM);
        float qv[32];
#pragma unroll
        for (int i = 0; i < 4; i++) {
            u16x8 t = q8[i];
#pragma unroll
            for (int j = 0; j < 8; j++) qv[i * 8 + j] = bf2f(t[j]);
        }
        float dots[4];
#pragma unroll
        for (int g = 0; g < 4; g++) {
            float acc = 0.0f;
#pragma unroll
            for (int i = 0; i < 4; i++) {
                u16x8 t = k8[g * 4 + i];
#pragma unroll
                for (int j = 0; j < 8; j++) acc += qv[i * 8 + j] * bf2f(t[j]);
            }
            dots[g] = acc * ATT_SCALE;
        }
        float mx = fmaxf(fmaxf(dots[0], dots[1]), fmaxf(dots[2], dots[3]));
        float e0 = __expf(dots[0] - mx), e1 = __expf(dots[1] - mx);
        float e2 = __expf(dots[2] - mx), e3 = __expf(dots[3] - mx);
        float inv = 1.0f / (e0 + e1 + e2 + e3);
        float a0 = e0 * inv, a1 = e1 * inv, a2 = e2 * inv, a3 = e3 * inv;
        ushort_t* orow = lds_x + p * XSTR + h * HD_DIM;   // pair tile is dead; reuse as attn-out
#pragma unroll
        for (int i = 0; i < 4; i++) {
            u16x8 t0 = v8[0 * 4 + i], t1 = v8[1 * 4 + i], t2 = v8[2 * 4 + i], t3 = v8[3 * 4 + i];
            u16x8 o;
#pragma unroll
            for (int j = 0; j < 8; j++) {
                float ov = a0 * bf2f(t0[j]) + a1 * bf2f(t1[j]) + a2 * bf2f(t2[j]) + a3 * bf2f(t3[j]);
                o[j] = f2bf(ov);
            }
            *(u16x8*)(orow + i * 8) = o;
        }
    }
    __syncthreads();

    // ---- phase 3: output projection (M=128, N=128, K=128) + bias, stores to HBM ----
    {
        v16bf a0 = load_a_lds(lds_x, wave * 16, 0,  XSTR);
        v16bf a1 = load_a_lds(lds_x, wave * 16, 32, XSTR);
        v16bf a2 = load_a_lds(lds_x, wave * 16, 64, XSTR);
        v16bf a3 = load_a_lds(lds_x, wave * 16, 96, XSTR);
        bool skipw = diag && (wave >= 4);    // diagonal: rows 64-127 are duplicates
#pragma unroll 2
        for (int nt = 0; nt < 8; nt++) {
            v8f c = {};
            c = wmma_bf16(a0, load_b_glb(wo_bf, nt * 16, 0),  c);
            c = wmma_bf16(a1, load_b_glb(wo_bf, nt * 16, 32), c);
            c = wmma_bf16(a2, load_b_glb(wo_bf, nt * 16, 64), c);
            c = wmma_bf16(a3, load_b_glb(wo_bf, nt * 16, 96), c);
            int n = nt * 16 + laneN;
            float bv = b_out[n];
            if (!skipw) {
#pragma unroll
                for (int r = 0; r < 8; r++) {
                    int p  = wave * 16 + halfw * 8 + r;
                    int rr = (p >> 3) & 7, cc = p & 7;
                    int pi = (p < 64) ? (i0 + rr) : (j0 + rr);
                    int pj = (p < 64) ? (j0 + cc) : (i0 + cc);
                    out[(((size_t)(b * S_DIM + pi)) * S_DIM + pj) * D_DIM + n] = c[r] + bv;
                }
            }
        }
    }
}

extern "C" void kernel_launch(void* const* d_in, const int* in_sizes, int n_in,
                              void* d_out, int out_size, void* d_ws, size_t ws_size,
                              hipStream_t stream) {
    const float* pair  = (const float*)d_in[0];
    const float* w_qkv = (const float*)d_in[1];
    const float* w_out = (const float*)d_in[2];
    const float* b_out = (const float*)d_in[3];
    ushort_t* wbf = (ushort_t*)d_ws;   // [0,49152): W_qkv bf16 ; [49152,65536): W_out bf16

    tri_convert_weights<<<(3 * D_DIM * D_DIM + 255) / 256, 256, 0, stream>>>(w_qkv, w_out, wbf);

    size_t lds_bytes = (size_t)LDS_USHORTS * sizeof(ushort_t);   // 135168 B (<320KB/WGP)
    tri_attn_kernel<<<2 * NPAIR, 256, lds_bytes, stream>>>(
        pair, wbf, wbf + 3 * D_DIM * D_DIM, b_out, (float*)d_out);
}
